// EEALNet_55147380081050
// MI455X (gfx1250) — matrix operand order repaired and checked by reading, not compiled, not run on hardware
//
#include <hip/hip_runtime.h>
#include <hip/hip_bf16.h>
#include <math.h>

typedef __attribute__((ext_vector_type(16))) _Float16 v16h;
typedef __attribute__((ext_vector_type(8)))  float    v8f;

#define BN_EPS 1e-5f

// ---------------------------------------------------------------------------
// Weight pre-pack: OIHW f32 -> f16 A-fragments in exact CDNA5 WMMA layout.
// K ordered tap-major: k_global = tap*CIN + c.  Within a 32-wide K-step,
// A-matrix 16x32 f16 layout (ISA 7.12.2): lane<16 -> M=lane, K halves
// {0..7,16..23}; lane>=16 -> same M, K halves {8..15,24..31}.
// coutReal < COUT pads unused output rows with zeros (Cout=1 out-conv).
// ---------------------------------------------------------------------------
template<int CIN, int COUT, int KSZ>
__global__ void pack_weights(const float* __restrict__ w, _Float16* __restrict__ ap,
                             int coutReal)
{
    constexpr int MT = COUT / 16;
    constexpr int KS = CIN / 32;
    const size_t total = (size_t)KSZ * KSZ * KS * MT * 512;
    size_t idx = (size_t)blockIdx.x * blockDim.x + threadIdx.x;
    if (idx >= total) return;
    const int i    = (int)(idx & 15);
    const int lane = (int)((idx >> 4) & 31);
    size_t rest = idx >> 9;
    const int mt = (int)(rest % MT); rest /= MT;
    const int ks = (int)(rest % KS); rest /= KS;
    const int t  = (int)rest;
    const int g  = lane >> 4;
    const int m  = mt * 16 + (lane & 15);
    const int kk = (i < 8) ? (g * 8 + i) : (16 + g * 8 + (i - 8));
    const int c  = ks * 32 + kk;
    const int ky = t / KSZ, kx = t % KSZ;
    float val = 0.f;
    if (m < coutReal)
        val = w[(((size_t)m * CIN + c) * KSZ + ky) * KSZ + kx];
    ap[idx] = (_Float16)val;
}

// ---------------------------------------------------------------------------
// Implicit-GEMM conv via v_wmma_f32_16x16x32_f16, f16 activations in/out.
// One wave -> MSUB 16(outch)x16(pixel) tiles sharing one B fragment:
// per K-step 16 d16 B-loads + 2*MSUB b128 A-loads feed MSUB WMMAs.
// OMODE: 0 bias, f16 store
//        1 bias+BN+ReLU, f16 store
//        2 bias+BN+ReLU+pixel-shuffle(2x), f16 store
//        3 channel 0 only (padded Cout), f32 store to [B,1,H,W]
//        4 bias+BN+ReLU, dual store: f32 (final output) + f16 (scratch)
// ---------------------------------------------------------------------------
template<int CIN, int COUT, int KSZ, int PAD, int OMODE, int MSUB>
__global__ __launch_bounds__(256)
void conv_wmma(const _Float16* __restrict__ in, const _Float16* __restrict__ apack,
               const float* __restrict__ bias, const float* __restrict__ bn,
               float* __restrict__ outf, _Float16* __restrict__ outh,
               int Bn, int Hh, int Ww)
{
    constexpr int MT = COUT / 16;
    constexpr int KS = CIN / 32;
    constexpr int MG = MT / MSUB;            // m-groups per output tile column
    const int wid = blockIdx.x * (blockDim.x >> 5) + (threadIdx.x >> 5);
    const int wt = Ww >> 4;
    const int ntiles = Bn * Hh * wt;
    if (wid >= MG * ntiles) return;          // whole-wave exit: EXEC all-1 at WMMA
    const int mg = wid % MG;                 // base m-tile = mg*MSUB
    const int nt = wid / MG;
    const int lane = threadIdx.x & 31;
    const int n = lane & 15, g = lane >> 4;
    const int b  = nt / (Hh * wt);
    const int r1 = nt % (Hh * wt);
    const int y  = r1 / wt;
    const int xn = (r1 % wt) * 16 + n;       // this lane's output x
    const size_t plane = (size_t)Hh * Ww;
    const size_t bbase = (size_t)b * CIN * plane;

    v8f acc[MSUB];
    #pragma unroll
    for (int ms = 0; ms < MSUB; ++ms) acc[ms] = (v8f){};

    for (int t = 0; t < KSZ * KSZ; ++t) {
        const int ky = t / KSZ, kx = t % KSZ;
        const int yy = y + ky - PAD;
        const int xx = xn + kx - PAD;
        const bool inb = (yy >= 0 && yy < Hh && xx >= 0 && xx < Ww);
        const size_t base = bbase + (size_t)(inb ? (yy * Ww + xx) : 0);

        // prefetch next tap's activation row (global_prefetch_b8)
        if (t + 1 < KSZ * KSZ) {
            const int t2 = t + 1;
            const int yy2 = y + t2 / KSZ - PAD;
            const int xx2 = xn + t2 % KSZ - PAD;
            if (yy2 >= 0 && yy2 < Hh && xx2 >= 0 && xx2 < Ww)
                __builtin_prefetch(in + bbase + (size_t)(g * 16) * plane
                                      + (size_t)yy2 * Ww + xx2, 0, 1);
        }

        #pragma unroll
        for (int ks = 0; ks < KS; ++ks) {
            // B-matrix 32x16 f16 layout: lane holds column n; lane group g
            // selects K: half i -> k = g*8 + i (i<8), 16 + g*8 + (i-8) (i>=8).
            const int c0 = ks * 32;
            v16h bf;
            #pragma unroll
            for (int i = 0; i < 16; ++i) {
                _Float16 v = (_Float16)0.f;
                if (inb) v = in[base + (size_t)(c0 + (i < 8 ? g * 8 + i : 8 + g * 8 + i)) * plane];
                bf[i] = v;
            }
            const size_t arow = ((size_t)t * KS + ks) * MT + mg * MSUB;
            #pragma unroll
            for (int ms = 0; ms < MSUB; ++ms) {
                const v16h af = *(const v16h*)(apack + ((arow + ms) * 32 + lane) * 16);
                acc[ms] = __builtin_amdgcn_wmma_f32_16x16x32_f16(
                    false, af, false, bf, (short)0, acc[ms], false, false);
            }
        }
    }

    // D layout: VGPR r -> M = r (lanes 0-15) / r+8 (lanes 16-31), N = lane&15
    #pragma unroll
    for (int ms = 0; ms < MSUB; ++ms) {
        const int mtile = mg * MSUB + ms;
        #pragma unroll
        for (int r = 0; r < 8; ++r) {
            const int m = mtile * 16 + r + 8 * g;
            if (OMODE == 3) {                // single real channel (padded Cout)
                if (m == 0)
                    outf[((size_t)b * Hh + y) * Ww + xn] = acc[ms][r] + bias[0];
                continue;
            }
            float alpha = 1.f;
            float ofs = bias ? bias[m] : 0.f;
            if (OMODE == 1 || OMODE == 2 || OMODE == 4) {
                // fold BN: y = conv*s + (bias-mean)*s + beta
                const float gg = bn[m], bb = bn[COUT + m];
                const float mm = bn[2 * COUT + m], vv = bn[3 * COUT + m];
                const float s = gg * rsqrtf(vv + BN_EPS);
                ofs = (ofs - mm) * s + bb;
                alpha = s;
            }
            float v = acc[ms][r] * alpha + ofs;
            if (OMODE == 1 || OMODE == 2 || OMODE == 4) v = fmaxf(v, 0.f);
            if (OMODE == 2) {                // pixel shuffle x2 at store
                const int c = m >> 2, dy = (m >> 1) & 1, dx = m & 1;
                outh[(((size_t)b * (COUT / 4) + c) * (2 * Hh) + (2 * y + dy)) * (2 * Ww)
                     + (2 * xn + dx)] = (_Float16)v;
            } else if (OMODE == 4) {         // final r: f32 out + f16 scratch
                const size_t o = (((size_t)b * COUT + m) * Hh + y) * Ww + xn;
                outf[o] = v;
                outh[o] = (_Float16)v;
            } else {
                outh[(((size_t)b * COUT + m) * Hh + y) * Ww + xn] = (_Float16)v;
            }
        }
    }
}

// ---------------------------------------------------------------------------
// Pointwise f16 staging kernels
// ---------------------------------------------------------------------------
__global__ void k_cvt_f16(const float* __restrict__ src, _Float16* __restrict__ dst,
                          int n)
{
    const int idx = blockIdx.x * blockDim.x + threadIdx.x;
    if (idx < n) dst[idx] = (_Float16)src[idx];
}

// out[b,c,p] = cur_x[b,c,p] * om[b,p]   (C=64, plane=128*128), f16
__global__ void k_mulmap_f16(const float* __restrict__ cur,
                             const float* __restrict__ om,
                             _Float16* __restrict__ dst)
{
    const int idx = blockIdx.x * blockDim.x + threadIdx.x;
    if (idx >= 8 * 64 * 128 * 128) return;
    const int pix = idx & 16383;
    const int b   = (idx >> 14) >> 6;
    dst[idx] = (_Float16)(cur[idx] * om[(b << 14) + pix]);
}

// e = dep + beta*fn (both f16 in, f32 math, f16 out)
__global__ void k_addbeta_f16(const _Float16* __restrict__ dep,
                              const _Float16* __restrict__ fn,
                              const float* __restrict__ betaPtr,
                              _Float16* __restrict__ e)
{
    const int idx = blockIdx.x * blockDim.x + threadIdx.x;
    if (idx >= 8 * 64 * 128 * 128) return;
    const float bv = betaPtr[0];
    e[idx] = (_Float16)((float)dep[idx] + bv * (float)fn[idx]);
}

// ---------------------------------------------------------------------------
// Map pipeline (tiny, pointwise). 8 x 128 x 128.
// ---------------------------------------------------------------------------
__global__ void k_maps_up(const float* __restrict__ in_map,
                          float* __restrict__ m0, float* __restrict__ m1)
{
    const int idx = blockIdx.x * blockDim.x + threadIdx.x;
    if (idx >= 8 * 128 * 128) return;
    const int x = idx & 127, y = (idx >> 7) & 127, b = idx >> 14;
    const float cy = y * (63.0f / 127.0f);
    const float cx = x * (63.0f / 127.0f);
    const int y0 = (int)floorf(cy), x0 = (int)floorf(cx);
    const int y1 = min(y0 + 1, 63), x1 = min(x0 + 1, 63);
    const float wy = cy - (float)y0, wx = cx - (float)x0;
    const float* p = in_map + (size_t)b * 64 * 64;
    const float v00 = p[y0 * 64 + x0], v01 = p[y0 * 64 + x1];
    const float v10 = p[y1 * 64 + x0], v11 = p[y1 * 64 + x1];
    const float u = (v00 * (1.f - wx) + v01 * wx) * (1.f - wy)
                  + (v10 * (1.f - wx) + v11 * wx) * wy;
    m0[idx] = 1.f / (1.f + expf(-u));               // sigmoid(up(in_map))
    m1[idx] = 1.f / (1.f + expf(-(1.f - u)));       // sigmoid(up(1-in_map))
}

__device__ __forceinline__ float max3x3(const float* __restrict__ s,
                                        int b, int y, int x)
{
    float mx = -__builtin_inff();
    for (int dy = -1; dy <= 1; ++dy) {
        const int yy = y + dy; if ((unsigned)yy >= 128u) continue;
        for (int dx = -1; dx <= 1; ++dx) {
            const int xx = x + dx; if ((unsigned)xx >= 128u) continue;
            mx = fmaxf(mx, s[((size_t)b << 14) + (yy << 7) + xx]);
        }
    }
    return mx;
}

__global__ void k_dilate_sub(const float* __restrict__ src, float* __restrict__ dst)
{
    const int idx = blockIdx.x * blockDim.x + threadIdx.x;
    if (idx >= 8 * 128 * 128) return;
    const int x = idx & 127, y = (idx >> 7) & 127, b = idx >> 14;
    dst[idx] = max3x3(src, b, y, x) - src[idx];
}

__global__ void k_combine(const float* __restrict__ inc0, const float* __restrict__ inc1,
                          const float* __restrict__ m1, float* __restrict__ om)
{
    const int idx = blockIdx.x * blockDim.x + threadIdx.x;
    if (idx >= 8 * 128 * 128) return;
    const int x = idx & 127, y = (idx >> 7) & 127, b = idx >> 14;
    const float inc2 = max3x3(m1, b, y, x) - m1[idx];
    om[idx] = inc0[idx] + inc1[idx] + inc2;
}

// ---------------------------------------------------------------------------
// Host-side orchestration
// ---------------------------------------------------------------------------
namespace {
constexpr size_t MAPB      = (size_t)8 * 128 * 128 * 4;      // 524288 B
constexpr size_t A16B      = (size_t)8 * 128 * 64 * 64 * 2;  // dep_x f16, 8 MB
constexpr size_t H16B      = (size_t)8 * 64 * 128 * 128 * 2; // 16 MB
constexpr size_t PACK_UP_H = (size_t)49 * 4 * 16 * 512;      // halves
constexpr size_t PACK_3_H  = (size_t)9 * 2 * 4 * 512;        // halves
constexpr size_t PACK_O_H  = (size_t)49 * 2 * 1 * 512;       // halves (out conv)
constexpr size_t OFF_M0   = 0;
constexpr size_t OFF_M1   = OFF_M0 + MAPB;
constexpr size_t OFF_I0   = OFF_M1 + MAPB;
constexpr size_t OFF_I1   = OFF_I0 + MAPB;
constexpr size_t OFF_OM   = OFF_I1 + MAPB;
constexpr size_t OFF_A16  = OFF_OM + MAPB;
constexpr size_t OFF_B0   = OFF_A16 + A16B;    // rotating f16 buffer 0 ("dep")
constexpr size_t OFF_B1   = OFF_B0 + H16B;     // rotating f16 buffer 1 ("fn")
constexpr size_t OFF_B2   = OFF_B1 + H16B;     // rotating f16 buffer 2 ("cur/e/r16")
constexpr size_t OFF_PUP  = OFF_B2 + H16B;
constexpr size_t OFF_PC2  = OFF_PUP + PACK_UP_H * 2;
constexpr size_t OFF_PD1  = OFF_PC2 + PACK_3_H * 2;
constexpr size_t OFF_PD2  = OFF_PD1 + PACK_3_H * 2;
constexpr size_t OFF_PD3  = OFF_PD2 + PACK_3_H * 2;
constexpr size_t OFF_POUT = OFF_PD3 + PACK_3_H * 2;
}

extern "C" void kernel_launch(void* const* d_in, const int* in_sizes, int n_in,
                              void* d_out, int out_size, void* d_ws, size_t ws_size,
                              hipStream_t stream)
{
    const float* cur_x  = (const float*)d_in[0];
    const float* dep_x  = (const float*)d_in[1];
    const float* in_map = (const float*)d_in[2];
    const float* up_w   = (const float*)d_in[3];
    const float* up_b   = (const float*)d_in[4];
    const float* up_bn  = (const float*)d_in[5];
    const float* c2_w   = (const float*)d_in[6];
    const float* c2_b   = (const float*)d_in[7];
    const float* d1_w   = (const float*)d_in[8];
    const float* d1_b   = (const float*)d_in[9];
    const float* d1_bn  = (const float*)d_in[10];
    const float* d2_w   = (const float*)d_in[11];
    const float* d2_b   = (const float*)d_in[12];
    const float* d2_bn  = (const float*)d_in[13];
    const float* d3_w   = (const float*)d_in[14];
    const float* d3_b   = (const float*)d_in[15];
    const float* d3_bn  = (const float*)d_in[16];
    const float* out_w  = (const float*)d_in[17];
    const float* out_b  = (const float*)d_in[18];
    const float* beta   = (const float*)d_in[19];

    char* ws = (char*)d_ws;
    float*     m0    = (float*)(ws + OFF_M0);
    float*     m1    = (float*)(ws + OFF_M1);
    float*     i0    = (float*)(ws + OFF_I0);
    float*     i1    = (float*)(ws + OFF_I1);
    float*     om    = (float*)(ws + OFF_OM);
    _Float16*  a16   = (_Float16*)(ws + OFF_A16);  // dep_x as f16
    _Float16*  b0    = (_Float16*)(ws + OFF_B0);
    _Float16*  b1    = (_Float16*)(ws + OFF_B1);
    _Float16*  b2    = (_Float16*)(ws + OFF_B2);
    _Float16*  p_up  = (_Float16*)(ws + OFF_PUP);
    _Float16*  p_c2  = (_Float16*)(ws + OFF_PC2);
    _Float16*  p_d1  = (_Float16*)(ws + OFF_PD1);
    _Float16*  p_d2  = (_Float16*)(ws + OFF_PD2);
    _Float16*  p_d3  = (_Float16*)(ws + OFF_PD3);
    _Float16*  p_out = (_Float16*)(ws + OFF_POUT);

    float* r_out  = (float*)d_out;               // [8,64,128,128]
    float* om_out = (float*)d_out + 8 * 64 * 128 * 128;

    // weight pre-pack (f32 -> f16 WMMA A-fragment layout)
    pack_weights<128, 256, 7><<<(unsigned)((PACK_UP_H + 255) / 256), 256, 0, stream>>>(up_w, p_up, 256);
    pack_weights< 64,  64, 3><<<(unsigned)((PACK_3_H  + 255) / 256), 256, 0, stream>>>(c2_w, p_c2, 64);
    pack_weights< 64,  64, 3><<<(unsigned)((PACK_3_H  + 255) / 256), 256, 0, stream>>>(d1_w, p_d1, 64);
    pack_weights< 64,  64, 3><<<(unsigned)((PACK_3_H  + 255) / 256), 256, 0, stream>>>(d2_w, p_d2, 64);
    pack_weights< 64,  64, 3><<<(unsigned)((PACK_3_H  + 255) / 256), 256, 0, stream>>>(d3_w, p_d3, 64);
    pack_weights< 64,  16, 7><<<(unsigned)((PACK_O_H  + 255) / 256), 256, 0, stream>>>(out_w, p_out, 1);

    // map pipeline
    k_maps_up   <<<512, 256, 0, stream>>>(in_map, m0, m1);
    k_dilate_sub<<<512, 256, 0, stream>>>(m0, i0);
    k_dilate_sub<<<512, 256, 0, stream>>>(i0, i1);
    k_combine   <<<512, 256, 0, stream>>>(i0, i1, m1, om);

    // stage dep_x as f16 (read 49x by up conv)
    k_cvt_f16<<<16384, 256, 0, stream>>>(dep_x, a16, 8 * 128 * 64 * 64);

    // up conv 7x7 128->256 @64x64, BN+ReLU+pixel-shuffle -> b0 = dep16 [8,64,128,128]
    conv_wmma<128, 256, 7, 3, 2, 4><<<1024, 256, 0, stream>>>(
        a16, p_up, up_b, up_bn, nullptr, b0, 8, 64, 64);

    // cur16 = outmap * cur_x (f16) -> b2
    k_mulmap_f16<<<32768, 256, 0, stream>>>(cur_x, om, b2);

    // conv2 3x3 64->64 @128x128 -> b1 = fn16
    conv_wmma<64, 64, 3, 1, 0, 4><<<1024, 256, 0, stream>>>(
        b2, p_c2, c2_b, nullptr, nullptr, b1, 8, 128, 128);

    // e16 = dep16 + beta*fn16 -> b2 (cur16 dead)
    k_addbeta_f16<<<32768, 256, 0, stream>>>(b0, b1, beta, b2);

    // d1: e16 -> b0 (dep16 dead), BN+ReLU
    conv_wmma<64, 64, 3, 1, 1, 4><<<1024, 256, 0, stream>>>(
        b2, p_d1, d1_b, d1_bn, nullptr, b0, 8, 128, 128);

    // d2: b0 -> b1 (fn16 dead), BN+ReLU
    conv_wmma<64, 64, 3, 1, 1, 4><<<1024, 256, 0, stream>>>(
        b0, p_d2, d2_b, d2_bn, nullptr, b1, 8, 128, 128);

    // d3: b1 -> r (dual store: f32 to d_out, f16 to b2), BN+ReLU
    conv_wmma<64, 64, 3, 1, 4, 4><<<1024, 256, 0, stream>>>(
        b1, p_d3, d3_b, d3_bn, r_out, b2, 8, 128, 128);

    // output_map = conv7x7 64->1 on r16, via WMMA with Cout padded to 16
    conv_wmma<64, 16, 7, 3, 3, 1><<<1024, 256, 0, stream>>>(
        b2, p_out, out_b, nullptr, om_out, nullptr, 8, 128, 128);
}
